// LSTMAttention_9740985828131
// MI455X (gfx1250) — matrix operand order repaired and checked by reading, not compiled, not run on hardware
//
#include <hip/hip_runtime.h>
#include <hip/hip_bf16.h>

// ---------------- CDNA5 WMMA types ----------------
typedef __bf16 bf16;
typedef __attribute__((ext_vector_type(16))) __bf16 v16bf;
typedef __attribute__((ext_vector_type(8)))  float  v8f;
// TDM descriptor vector types (6-arg builtin: v4u, v8i, v4i, v4i, v8i, i32)
typedef unsigned int v4u __attribute__((ext_vector_type(4)));
typedef int          v8i __attribute__((ext_vector_type(8)));
typedef int          v4i __attribute__((ext_vector_type(4)));

// Problem constants (from reference)
constexpr int Bn = 64;     // batch
constexpr int Tn = 256;    // time steps
constexpr int In = 512;    // input features
constexpr int Hn = 1024;   // hidden
constexpr int Cn = 1024;   // ctx features
constexpr int Sn = 256;    // ctx length

__device__ __forceinline__ float fast_sigmoid(float x) {
    return 1.0f / (1.0f + __expf(-x));
}
__device__ __forceinline__ float fast_tanh(float x) {
    float e = __expf(2.0f * x);
    return 1.0f - 2.0f / (e + 1.0f);
}

// ---- A fragment: 16x32 (MxK) bf16, contiguous k order: element e of lane l holds k = 16*(l/16) + e (as in the B packing).
__device__ __forceinline__ v16bf load_a_frag(const float* __restrict__ base,
                                             int rowStride, int m0, int kbase, int lane) {
    const float* p = base + (long)(m0 + (lane & 15)) * rowStride + kbase + ((lane >> 4) << 4);
    v16bf a;
#pragma unroll
    for (int e = 0; e < 8; ++e) a[e] = (bf16)p[e];
#pragma unroll
    for (int e = 0; e < 8; ++e) a[8 + e] = (bf16)p[8 + e];
    return a;
}

// ---- B fragment: 32x16 (KxN) bf16 packed fragment-native tiles of 512 bf16.
__device__ __forceinline__ v16bf load_b_frag(const bf16* __restrict__ packed, long tile, int lane) {
    return *(const v16bf*)(packed + tile * 512 + lane * 16);
}

__device__ __forceinline__ v8f wmma_bf16(v16bf a, v16bf b, v8f c) {
    return __builtin_amdgcn_wmma_f32_16x16x32_bf16(false, a, false, b, (short)0, c, false, false);
}

// ---------------- Weight packing: fp32 -> bf16 WMMA-B tiles ----------------
__global__ void pack_b_kernel(const float* __restrict__ src, bf16* __restrict__ dst,
                              int K, int N, int transpose) {
    long idx = (long)blockIdx.x * blockDim.x + threadIdx.x;
    long total = (long)K * N;
    if (idx >= total) return;
    int  n = (int)(idx % N);
    int  k = (int)(idx / N);
    float v = transpose ? src[(long)n * K + k] : src[(long)k * N + n];
    int kt = k >> 5, kk = k & 31, nt = n >> 4, nn = n & 15;
    int lane = nn + ((kk >> 4) << 4);
    int e = kk & 15;
    long tile = (long)nt * (K >> 5) + kt;
    dst[tile * 512 + (long)lane * 16 + e] = (bf16)v;
}

// ---------------- projected_ctx = ctx @ c2a + b_c2a ----------------
__global__ void pc_gemm_kernel(const float* __restrict__ A, const bf16* __restrict__ Bp,
                               const float* __restrict__ bias, float* __restrict__ Out) {
    int lane = threadIdx.x & 31;
    int wid  = blockIdx.x * (blockDim.x >> 5) + (threadIdx.x >> 5);
    int mt = wid & 1023;   // M/16 = 1024 row tiles
    int nt = wid >> 10;    // 64 col tiles
    int m0 = mt << 4, n0 = nt << 4;
    v8f acc = {0.f, 0.f, 0.f, 0.f, 0.f, 0.f, 0.f, 0.f};
    for (int kt = 0; kt < 32; ++kt) {
        v16bf a = load_a_frag(A, Cn, m0, kt << 5, lane);
        acc = wmma_bf16(a, load_b_frag(Bp, (long)nt * 32 + kt, lane), acc);
    }
    int col = n0 + (lane & 15);
    float bv = bias[col];
    int rbase = m0 + ((lane >> 4) << 3);
#pragma unroll
    for (int v = 0; v < 8; ++v)
        Out[(long)(rbase + v) * Cn + col] = acc[v] + bv;
}

// ---------------- Fused LSTM step: gates GEMM (two K segments) + cell ----------------
__global__ void lstm_step_kernel(const float* __restrict__ xbase, int xstride, int kx,
                                 const float* __restrict__ hprev,
                                 const float* __restrict__ cprev,
                                 const bf16* __restrict__ xwp,   // packed (kx, 4H)
                                 const bf16* __restrict__ hwp,   // packed (H, 4H)
                                 const float* __restrict__ ib,
                                 const float* __restrict__ hb,
                                 float* __restrict__ hout, float* __restrict__ cout) {
    int lane = threadIdx.x & 31;
    int wid  = blockIdx.x * (blockDim.x >> 5) + (threadIdx.x >> 5);
    int mt = wid & 3;    // B/16 = 4
    int nt = wid >> 2;   // H/16 = 64
    int m0 = mt << 4, j0 = nt << 4;
    v8f aI = {0.f,0.f,0.f,0.f,0.f,0.f,0.f,0.f};
    v8f aF = aI, aG = aI, aO = aI;

    int kxt = kx >> 5;
    for (int kt = 0; kt < kxt; ++kt) {          // x-input segment
        v16bf a = load_a_frag(xbase, xstride, m0, kt << 5, lane);
        aI = wmma_bf16(a, load_b_frag(xwp, (long)(0 * 64 + nt) * kxt + kt, lane), aI);
        aF = wmma_bf16(a, load_b_frag(xwp, (long)(1 * 64 + nt) * kxt + kt, lane), aF);
        aG = wmma_bf16(a, load_b_frag(xwp, (long)(2 * 64 + nt) * kxt + kt, lane), aG);
        aO = wmma_bf16(a, load_b_frag(xwp, (long)(3 * 64 + nt) * kxt + kt, lane), aO);
    }
    for (int kt = 0; kt < Hn / 32; ++kt) {      // hidden segment (K=1024)
        v16bf a = load_a_frag(hprev, Hn, m0, kt << 5, lane);
        aI = wmma_bf16(a, load_b_frag(hwp, (long)(0 * 64 + nt) * 32 + kt, lane), aI);
        aF = wmma_bf16(a, load_b_frag(hwp, (long)(1 * 64 + nt) * 32 + kt, lane), aF);
        aG = wmma_bf16(a, load_b_frag(hwp, (long)(2 * 64 + nt) * 32 + kt, lane), aG);
        aO = wmma_bf16(a, load_b_frag(hwp, (long)(3 * 64 + nt) * 32 + kt, lane), aO);
    }

    int col   = j0 + (lane & 15);
    int rbase = m0 + ((lane >> 4) << 3);
    float bI = ib[0 * Hn + col] + hb[0 * Hn + col];
    float bF = ib[1 * Hn + col] + hb[1 * Hn + col];
    float bG = ib[2 * Hn + col] + hb[2 * Hn + col];
    float bO = ib[3 * Hn + col] + hb[3 * Hn + col];
#pragma unroll
    for (int v = 0; v < 8; ++v) {
        int b = rbase + v;
        float gi = aI[v] + bI, gf = aF[v] + bF, gg = aG[v] + bG, go = aO[v] + bO;
        float cp = cprev[(long)b * Hn + col];
        float cc = fast_sigmoid(gf) * cp + fast_sigmoid(gi) * fast_tanh(gg);
        float hh = fast_sigmoid(go) * fast_tanh(cc);
        cout[(long)b * Hn + col] = cc;
        hout[(long)b * Hn + col] = hh;
    }
}

// ---------------- hcpi = h1 @ h2a.T + x_t @ i2a  (fused, no bias) ----------------
__global__ void hcpi_kernel(const float* __restrict__ h1,
                            const float* __restrict__ xbase, int xstride,
                            const bf16* __restrict__ h2ap,   // packed (H, C)
                            const bf16* __restrict__ i2ap,   // packed (I, C)
                            float* __restrict__ out) {
    int lane = threadIdx.x & 31;
    int wid  = blockIdx.x * (blockDim.x >> 5) + (threadIdx.x >> 5);
    int mt = wid & 3;    // 4
    int nt = wid >> 2;   // 64
    int m0 = mt << 4, n0 = nt << 4;
    v8f acc = {0.f,0.f,0.f,0.f,0.f,0.f,0.f,0.f};
    for (int kt = 0; kt < Hn / 32; ++kt) {
        v16bf a = load_a_frag(h1, Hn, m0, kt << 5, lane);
        acc = wmma_bf16(a, load_b_frag(h2ap, (long)nt * 32 + kt, lane), acc);
    }
    for (int kt = 0; kt < In / 32; ++kt) {
        v16bf a = load_a_frag(xbase, xstride, m0, kt << 5, lane);
        acc = wmma_bf16(a, load_b_frag(i2ap, (long)nt * 16 + kt, lane), acc);
    }
    int col = n0 + (lane & 15);
    int rbase = m0 + ((lane >> 4) << 3);
#pragma unroll
    for (int v = 0; v < 8; ++v)
        out[(long)(rbase + v) * Cn + col] = acc[v];
}

// ---------------- Attention with TDM-staged pc sweep ----------------
// Chunking: KC=16 columns per chunk; TDM copies a 2D tile (16 cols x 256 rows,
// row stride = B*C elems) into LDS with 1-DWORD row padding (17 % 64, gcd=1 =>
// bank-conflict-free reads at fixed j across threads).
constexpr int KC  = 16;            // columns per chunk
constexpr int KCP = KC + 1;        // padded row pitch (dwords)
constexpr int NCHUNK = Cn / KC;    // 64 chunks

struct AttnLds {
    float hbuf[Cn];
    float rbuf[Cn];
    float red[Sn];
    float abuf[Sn];
    float stage[2][Sn * KCP];      // 2 x 17408 B
};

__device__ __forceinline__ unsigned lds_byte_addr(const void* p) {
    // generic LDS address = aperture(hi32) | lds_offset(lo32)
    return (unsigned)(unsigned long long)p;
}

// Issue one TDM 2D tile load: pc[(s*Bn+b), cc*KC .. +KC-1] for s=0..Sn-1 -> LDS
__device__ __forceinline__ void tdm_load_pc_chunk(const float* __restrict__ pc,
                                                  int b, int cc, unsigned ldsAddr) {
    unsigned long long ga = (unsigned long long)(const char*)pc
                          + (unsigned long long)b * (Cn * 4ull)
                          + (unsigned long long)cc * (KC * 4ull);
    v4u g0;
    g0[0] = 1u;                                   // count=1, user descriptor
    g0[1] = ldsAddr;                              // lds_addr
    g0[2] = (unsigned)ga;                         // global_addr[31:0]
    g0[3] = (unsigned)((ga >> 32) & 0x1FFFFFFull) // global_addr[56:32]
          | (2u << 30);                           // type=2 (image)
    v8i g1;
    g1[0] = (int)((2u << 16)     // data_size = 4B
                | (1u << 20)     // pad_enable
                | (3u << 22)     // pad_interval code 3 = 16 DWORDs
                | (0u << 25));   // pad_amount  code 0 = 1 DWORD
    g1[1] = 0;                                    // atomic barrier addr (unused), tensor_dim0 lo16=0
    g1[2] = 16;                                   // tensor_dim0 = 1<<20 (hi16), tensor_dim1 lo16=0
    g1[3] = (int)(16u | ((unsigned)KC << 16));    // tensor_dim1 hi16, tile_dim0 = KC
    g1[4] = Sn;                                   // tile_dim1 = 256, tile_dim2 = 0
    g1[5] = Bn * Cn;                              // tensor_dim0_stride = 65536 elems (lo32)
    g1[6] = 0;
    g1[7] = 0;
    v4i gz4 = {0, 0, 0, 0};                       // groups 2/3: tile_dim3/4 = 0 (unused)
    v8i gz8 = {0, 0, 0, 0, 0, 0, 0, 0};
    __builtin_amdgcn_tensor_load_to_lds(g0, g1, gz4, gz4, gz8, 0);
}

__global__ void attn_kernel(const float* __restrict__ pc,    // (S*B, C)
                            const float* __restrict__ hcpi,  // (B, C)
                            const float* __restrict__ r2a,   // (C)
                            const float* __restrict__ ctx,   // (S*B, C)
                            float* __restrict__ wctx) {      // (B, C)
    __shared__ AttnLds L;
    int b = blockIdx.x, tid = threadIdx.x;
    bool issuer = (tid < 32);                      // wave 0 drives the TDM

    if (issuer) tdm_load_pc_chunk(pc, b, 0, lds_byte_addr(&L.stage[0][0]));
    for (int c = tid; c < Cn; c += Sn) {           // overlapped with TDM
        L.hbuf[c] = hcpi[(long)b * Cn + c];
        L.rbuf[c] = r2a[c];
    }
    if (issuer) __builtin_amdgcn_s_wait_tensorcnt(0);
    __syncthreads();

    // score for s = tid: sum_c tanh(pc[s,b,c] + hcpi[b,c]) * r[c]
    float acc = 0.f;
    for (int cc = 0; cc < NCHUNK; ++cc) {
        int p = cc & 1;
        if (issuer && (cc + 1 < NCHUNK))
            tdm_load_pc_chunk(pc, b, cc + 1, lds_byte_addr(&L.stage[1 - p][0]));
        const float* row = &L.stage[p][tid * KCP];
        const float* hb  = &L.hbuf[cc * KC];
        const float* rb  = &L.rbuf[cc * KC];
#pragma unroll
        for (int j = 0; j < KC; ++j)
            acc += fast_tanh(row[j] + hb[j]) * rb[j];
        if (issuer) __builtin_amdgcn_s_wait_tensorcnt(0);
        __syncthreads();
    }

    // softmax over s
    L.red[tid] = acc;
    __syncthreads();
    for (int off = Sn / 2; off > 0; off >>= 1) {
        if (tid < off) L.red[tid] = fmaxf(L.red[tid], L.red[tid + off]);
        __syncthreads();
    }
    float mx = L.red[0];
    __syncthreads();
    float e = __expf(acc - mx);
    L.red[tid] = e;
    __syncthreads();
    for (int off = Sn / 2; off > 0; off >>= 1) {
        if (tid < off) L.red[tid] += L.red[tid + off];
        __syncthreads();
    }
    float denom = L.red[0];
    __syncthreads();
    L.abuf[tid] = e / denom;
    __syncthreads();

    // wctx[b,c] = sum_s ctx[s,b,c] * alpha[s]  (float4, coalesced over c)
    {
        const float4* cb = (const float4*)(ctx + (long)b * Cn) + tid;  // 4 cols per thread
        float4 w = {0.f, 0.f, 0.f, 0.f};
        for (int s2 = 0; s2 < Sn; ++s2) {
            float4 v = cb[(long)s2 * (Bn * Cn / 4)];
            float a = L.abuf[s2];
            w.x += v.x * a; w.y += v.y * a; w.z += v.z * a; w.w += v.w * a;
        }
        ((float4*)(wctx + (long)b * Cn))[tid] = w;
    }
}

// ---------------- Host side ----------------
extern "C" void kernel_launch(void* const* d_in, const int* in_sizes, int n_in,
                              void* d_out, int out_size, void* d_ws, size_t ws_size,
                              hipStream_t stream) {
    const float* input = (const float*)d_in[0];   // (B,T,I)
    const float* hx    = (const float*)d_in[1];   // (B,H)
    const float* cx    = (const float*)d_in[2];   // (B,H)
    const float* ctx   = (const float*)d_in[3];   // (S,B,C)
    const float* iw1   = (const float*)d_in[4];   // (4H,I)
    const float* hw1   = (const float*)d_in[5];   // (4H,H)
    const float* ib1   = (const float*)d_in[6];
    const float* hb1   = (const float*)d_in[7];
    const float* iw2   = (const float*)d_in[8];   // (4H,C)
    const float* hw2   = (const float*)d_in[9];   // (4H,H)
    const float* ib2   = (const float*)d_in[10];
    const float* hb2   = (const float*)d_in[11];
    const float* c2a   = (const float*)d_in[12];  // (C,C)
    const float* b_c2a = (const float*)d_in[13];
    const float* h2a   = (const float*)d_in[14];  // (C,H)
    const float* i2a   = (const float*)d_in[15];  // (I,C)
    const float* r2a   = (const float*)d_in[16];  // (C,1)

    char* ws = (char*)d_ws;
    size_t off = 0;
    auto alloc = [&](size_t bytes) {
        char* p = ws + off;
        off = (off + bytes + 255) & ~(size_t)255;
        return p;
    };
    bf16* iw1p = (bf16*)alloc((size_t)In * 4 * Hn * 2);
    bf16* hw1p = (bf16*)alloc((size_t)Hn * 4 * Hn * 2);
    bf16* iw2p = (bf16*)alloc((size_t)Cn * 4 * Hn * 2);
    bf16* hw2p = (bf16*)alloc((size_t)Hn * 4 * Hn * 2);
    bf16* h2ap = (bf16*)alloc((size_t)Hn * Cn * 2);
    bf16* i2ap = (bf16*)alloc((size_t)In * Cn * 2);
    bf16* c2ap = (bf16*)alloc((size_t)Cn * Cn * 2);
    float* pc     = (float*)alloc((size_t)Sn * Bn * Cn * 4);
    float* h_mid  = (float*)alloc((size_t)Bn * Hn * 4);
    float* c_mid  = (float*)alloc((size_t)Bn * Hn * 4);
    float* c_carry= (float*)alloc((size_t)Bn * Hn * 4);
    float* hcpi   = (float*)alloc((size_t)Bn * Cn * 4);
    float* wctx   = (float*)alloc((size_t)Bn * Cn * 4);
    (void)ws_size; (void)in_sizes; (void)n_in; (void)out_size;

    auto pack = [&](const float* src, bf16* dst, int K, int N, int tr) {
        long total = (long)K * N;
        int blocks = (int)((total + 255) / 256);
        pack_b_kernel<<<blocks, 256, 0, stream>>>(src, dst, K, N, tr);
    };
    pack(iw1, iw1p, In, 4 * Hn, 1);
    pack(hw1, hw1p, Hn, 4 * Hn, 1);
    pack(iw2, iw2p, Cn, 4 * Hn, 1);
    pack(hw2, hw2p, Hn, 4 * Hn, 1);
    pack(h2a, h2ap, Hn, Cn, 1);
    pack(i2a, i2ap, In, Cn, 0);
    pack(c2a, c2ap, Cn, Cn, 0);

    // projected_ctx (one-time WMMA GEMM); 67MB result stays L2-resident (192MB L2)
    pc_gemm_kernel<<<(Sn * Bn / 16) * (Cn / 16) / 8, 256, 0, stream>>>(ctx, c2ap, b_c2a, pc);

    float* out = (float*)d_out;                // [T*B*H] outputs, then h_f, c_f

    for (int t = 0; t < Tn; ++t) {
        const float* x_t   = input + (long)t * In;               // rows stride T*I
        const float* hprev = t ? out + (long)(t - 1) * Bn * Hn : hx;
        const float* cprev = t ? c_carry : cx;

        lstm_step_kernel<<<64, 128, 0, stream>>>(x_t, Tn * In, In, hprev, cprev,
                                                 iw1p, hw1p, ib1, hb1, h_mid, c_mid);
        hcpi_kernel<<<64, 128, 0, stream>>>(h_mid, x_t, Tn * In, h2ap, i2ap, hcpi);
        attn_kernel<<<Bn, Sn, 0, stream>>>(pc, hcpi, r2a, ctx, wctx);
        lstm_step_kernel<<<64, 128, 0, stream>>>(wctx, Cn, Cn, h_mid, c_mid,
                                                 iw2p, hw2p, ib2, hb2,
                                                 out + (long)t * Bn * Hn, c_carry);
    }

    (void)hipMemcpyAsync(out + (size_t)Tn * Bn * Hn,
                         out + (size_t)(Tn - 1) * Bn * Hn,
                         (size_t)Bn * Hn * sizeof(float), hipMemcpyDeviceToDevice, stream);
    (void)hipMemcpyAsync(out + (size_t)Tn * Bn * Hn + (size_t)Bn * Hn,
                         c_carry,
                         (size_t)Bn * Hn * sizeof(float), hipMemcpyDeviceToDevice, stream);
}
